// MemoryEfficientAttention_49108656062985
// MI455X (gfx1250) — matrix operand order repaired and checked by reading, not compiled, not run on hardware
//
#include <hip/hip_runtime.h>

typedef __attribute__((ext_vector_type(16))) __bf16 v16bf;
typedef __attribute__((ext_vector_type(8)))  __bf16 v8bf;
typedef __attribute__((ext_vector_type(4)))  __bf16 v4bf;
typedef __attribute__((ext_vector_type(8)))  float  v8f;

#define EMBED 1024
#define SEQ   4096
#define HEADS 16
#define HDIM  64

__device__ __forceinline__ v8f wmma_bf16(v16bf a, v16bf b, v8f c) {
  // D = A(16x32 bf16) * B(32x16 bf16) + C(16x16 f32)
  return __builtin_amdgcn_wmma_f32_16x16x32_bf16(
      /*neg_a=*/false, a, /*neg_b=*/false, b,
      /*c_mod=*/(short)0, c, /*reuse_a=*/false, /*reuse_b=*/false);
}

__device__ __forceinline__ v16bf make_a(v8bf lo, v8bf hi) {
  return __builtin_shufflevector(lo, hi, 0,1,2,3,4,5,6,7,8,9,10,11,12,13,14,15);
}

// A-fragment 16x32 from row-major [*, lda] at (row0, k0).
// ISA layout: lanes 0-15 -> M=lane, K chunks {k0..k0+7, k0+16..k0+23};
//             lanes 16-31 -> M=lane-16, K chunks {k0+8..+15, k0+24..+31}.
__device__ __forceinline__ v16bf load_a_frag(const __bf16* base, int lda,
                                             int row0, int k0, int lane) {
  const int r  = row0 + (lane & 15);
  const int kb = k0 + ((lane < 16) ? 0 : 8);
  const __bf16* p = base + (size_t)r * lda + kb;
  return make_a(*reinterpret_cast<const v8bf*>(p),
                *reinterpret_cast<const v8bf*>(p + 16));
}

// B-fragment 32x16 where each B column n is a contiguous row of a row-major
// [N, K] array (global or LDS). ISA layout: lane%16 = column; lanes 0-15 hold
// K chunks {k0..k0+7, k0+16..k0+23}, lanes 16-31 hold {k0+8..+15, k0+24..+31}.
__device__ __forceinline__ v16bf load_b_frag(const __bf16* base, int ldb,
                                             int col0, int k0, int lane) {
  const int n  = col0 + (lane & 15);
  const int kb = k0 + ((lane < 16) ? 0 : 8);
  const __bf16* p = base + (size_t)n * ldb + kb;
  return make_a(*reinterpret_cast<const v8bf*>(p),
                *reinterpret_cast<const v8bf*>(p + 16));
}

// Cooperative async DMA of one A tile (128 x 64) and one B tile (128 x 64)
// into LDS (rows padded to 72 elems / 144B for bank-conflict-free fragment
// reads). 256 threads x 4 chunks x 16B per tile; 8 async ops per wave.
__device__ __forceinline__ void stage_ab(unsigned long long gA,
                                         unsigned long long gB,
                                         int rowA0, int rowB0, int k0,
                                         unsigned ldsA, unsigned ldsB, int tid) {
#pragma unroll
  for (int j = 0; j < 4; ++j) {
    const int c = tid + 256 * j;                  // chunk 0..1023
    const int row = c >> 3, sub = c & 7;          // 8 x 16B chunks per row
    unsigned lo = ldsA + (unsigned)(row * 144 + sub * 16);
    unsigned go = (unsigned)(((rowA0 + row) * EMBED + k0 + sub * 8) * 2);
    asm volatile("global_load_async_to_lds_b128 %0, %1, %2"
                 :: "v"(lo), "v"(go), "s"(gA) : "memory");
    lo = ldsB + (unsigned)(row * 144 + sub * 16);
    go = (unsigned)(((rowB0 + row) * EMBED + k0 + sub * 8) * 2);
    asm volatile("global_load_async_to_lds_b128 %0, %1, %2"
                 :: "v"(lo), "v"(go), "s"(gB) : "memory");
  }
}

// ---------------------------------------------------------------- fp32 -> bf16
__global__ void cvt_f32_bf16(const float* __restrict__ in,
                             __bf16* __restrict__ out, int n) {
  int i = (blockIdx.x * blockDim.x + threadIdx.x) * 4;
  if (i < n) {
    float4 f = *reinterpret_cast<const float4*>(in + i);
    v4bf o;
    o[0] = (__bf16)f.x; o[1] = (__bf16)f.y; o[2] = (__bf16)f.z; o[3] = (__bf16)f.w;
    *reinterpret_cast<v4bf*>(out + i) = o;
  }
}

// ------------------------------------------------- QKV projection + scatter
// qkv = x @ w_qkv^T ; scatter into Q[h][t][d], K[h][t][d], Vt[h][d][t]
// Block = 128x128 tile (8 waves x 16x128); A/B tiles double-buffered in LDS
// via async DMA; inner loop is LDS->WMMA only.
__global__ void qkv_gemm(const __bf16* __restrict__ xb,
                         const __bf16* __restrict__ wb,
                         __bf16* __restrict__ Q, __bf16* __restrict__ K,
                         __bf16* __restrict__ Vt) {
  const int tid  = threadIdx.x;
  const int lane = tid & 31;
  const int wave = tid >> 5;
  const int NT   = (3 * EMBED) / 128;              // 24 n-tiles
  const int bt0  = (blockIdx.x / NT) * 128;        // block M origin
  const int f0   = (blockIdx.x % NT) * 128;        // block N origin
  const int t0   = bt0 + wave * 16;                // wave M origin

  __shared__ __bf16 sA[2][128 * 72];               // 2 x 18KB (padded rows)
  __shared__ __bf16 sB[2][128 * 72];               // 2 x 18KB
  const unsigned aA[2] = { (unsigned)(size_t)&sA[0][0], (unsigned)(size_t)&sA[1][0] };
  const unsigned aB[2] = { (unsigned)(size_t)&sB[0][0], (unsigned)(size_t)&sB[1][0] };
  const unsigned long long gA = (unsigned long long)(size_t)xb;
  const unsigned long long gB = (unsigned long long)(size_t)wb;

  v8f acc[8] = {};
  const int NSTEP = EMBED / 64;                    // 16 K-steps of 64
  stage_ab(gA, gB, bt0, f0, 0, aA[0], aB[0], tid);
  for (int it = 0; it < NSTEP; ++it) {
    const int cur = it & 1;
    const int nk  = (it + 1 < NSTEP) ? (it + 1) * 64 : it * 64;  // clamp (dup)
    stage_ab(gA, gB, bt0, f0, nk, aA[cur ^ 1], aB[cur ^ 1], tid);
    asm volatile("s_wait_asynccnt 0x8" ::: "memory"); // current tile DMA done
    __syncthreads();
    const __bf16* sa = &sA[cur][0];
    const __bf16* sb = &sB[cur][0];
#pragma unroll
    for (int kd = 0; kd < 2; ++kd) {
      v16bf a = load_a_frag(sa, 72, wave * 16, kd * 32, lane);
#pragma unroll
      for (int n = 0; n < 8; ++n) {
        v16bf b = load_b_frag(sb, 72, n * 16, kd * 32, lane);
        acc[n] = wmma_bf16(a, b, acc[n]);
      }
    }
    __syncthreads();          // done reading buf[cur] before it is rewritten
  }

  const int col   = lane & 15;
  const int rbase = (lane < 16) ? 0 : 8;
#pragma unroll
  for (int n = 0; n < 8; ++n) {
    const int f = f0 + n * 16 + col;
#pragma unroll
    for (int i = 0; i < 8; ++i) {
      const int t = t0 + rbase + i;
      const __bf16 v = (__bf16)acc[n][i];
      if (f < EMBED) {
        const int h = f >> 6, d = f & 63;
        Q[((size_t)h * SEQ + t) * HDIM + d] = v;
      } else if (f < 2 * EMBED) {
        const int f2 = f - EMBED, h = f2 >> 6, d = f2 & 63;
        K[((size_t)h * SEQ + t) * HDIM + d] = v;
      } else {
        const int f2 = f - 2 * EMBED, h = f2 >> 6, d = f2 & 63;
        Vt[((size_t)h * HDIM + d) * SEQ + t] = v;   // transposed V
      }
    }
  }
}

// --------------------------------------------------------- flash attention
// Block = 8 waves x 16 queries, one head. K/V tiles (32 keys) staged into LDS
// once per block via async-to-LDS DMA, double-buffered to overlap with math.
// All LDS tiles row-padded for conflict-free fragment reads.
__global__ void flash_attn(const __bf16* __restrict__ Q,
                           const __bf16* __restrict__ Kc,
                           const __bf16* __restrict__ Vt,
                           __bf16* __restrict__ O) {
  const int tid  = threadIdx.x;
  const int lane = tid & 31;
  const int wave = tid >> 5;
  const int head = blockIdx.x >> 5;                 // 32 blocks per head
  const int q0   = ((blockIdx.x & 31) * 8 + wave) * 16;

  const __bf16* Qh = Q  + (size_t)head * SEQ * HDIM;
  const __bf16* Kh = Kc + (size_t)head * SEQ * HDIM;
  const __bf16* Vh = Vt + (size_t)head * HDIM * SEQ;  // [64][4096]

  __shared__ __bf16 sK[2][32 * 72];                 // K tile [32][64]+pad
  __shared__ __bf16 sV[2][HDIM * 40];               // V tile [64][32]+pad
  __shared__ __bf16 sP[8][16 * 40];                 // per-wave P staging +pad
  __bf16* P = sP[wave];

  const unsigned aK[2] = { (unsigned)(size_t)&sK[0][0], (unsigned)(size_t)&sK[1][0] };
  const unsigned aV[2] = { (unsigned)(size_t)&sV[0][0], (unsigned)(size_t)&sV[1][0] };
  const unsigned long long gK = (unsigned long long)(size_t)Kh;
  const unsigned long long gV = (unsigned long long)(size_t)Vh;

  // Each of the 256 threads owns one 16B chunk of each tile:
  // K tile chunk c: row=c>>3, sub=c&7 ; V tile chunk c: d=c>>2, sub=c&3.
  const int krow = tid >> 3, ksub = tid & 7;
  const int vrow = tid >> 2, vsub = tid & 3;

#define STAGE_TILES(kb, buf)                                                   \
  {                                                                            \
    unsigned goK = (unsigned)((((kb) + krow) * HDIM + ksub * 8) * 2);          \
    unsigned loK = aK[buf] + (unsigned)(krow * 144 + ksub * 16);               \
    asm volatile("global_load_async_to_lds_b128 %0, %1, %2"                    \
                 :: "v"(loK), "v"(goK), "s"(gK) : "memory");                   \
    unsigned goV = (unsigned)((vrow * SEQ + (kb) + vsub * 8) * 2);             \
    unsigned loV = aV[buf] + (unsigned)(vrow * 80 + vsub * 16);                \
    asm volatile("global_load_async_to_lds_b128 %0, %1, %2"                    \
                 :: "v"(loV), "v"(goV), "s"(gV) : "memory");                   \
  }

  v16bf qf[2];
  qf[0] = load_a_frag(Qh, HDIM, q0, 0, lane);
  qf[1] = load_a_frag(Qh, HDIM, q0, 32, lane);

  v8f o[4] = {};
  float m[8], l[8];
#pragma unroll
  for (int i = 0; i < 8; ++i) { m[i] = -1e30f; l[i] = 0.f; }

  const float sc = 0.125f;                          // 1/sqrt(64)
  const int rbase = (lane < 16) ? 0 : 8;
  const int col   = lane & 15;
  const int NIT   = SEQ / 32;                       // 128 key tiles

  STAGE_TILES(0, 0);                                // prologue: tile 0
  for (int it = 0; it < NIT; ++it) {
    const int cur = it & 1;
    const int nkb = (it + 1 < NIT) ? (it + 1) * 32 : it * 32;  // clamp (dup ok)
    STAGE_TILES(nkb, cur ^ 1);                      // prefetch next tile
    asm volatile("s_wait_asynccnt 0x2" ::: "memory"); // current tile DMA done
    __syncthreads();                                // ...block-wide
    const __bf16* kt = &sK[cur][0];                 // [32 keys][72]
    const __bf16* vt = &sV[cur][0];                 // [64 d][40]

    // S = Q @ K^T for 32 keys (two 16-key column tiles), K-dim = 64
    v16bf kf[4];
#pragma unroll
    for (int kd = 0; kd < 2; ++kd) {
      kf[kd * 2]     = load_b_frag(kt, 72, 0,  kd * 32, lane);
      kf[kd * 2 + 1] = load_b_frag(kt, 72, 16, kd * 32, lane);
    }
    v8f s0 = {}, s1 = {};
#pragma unroll
    for (int kd = 0; kd < 2; ++kd) {
      s0 = wmma_bf16(qf[kd], kf[kd * 2],     s0);
      s1 = wmma_bf16(qf[kd], kf[kd * 2 + 1], s1);
    }
    // online softmax: row stats replicated across the 16-lane halves
#pragma unroll
    for (int i = 0; i < 8; ++i) {
      float a0 = s0[i] * sc, a1 = s1[i] * sc;
      float cm = fmaxf(a0, a1);
      cm = fmaxf(cm, __shfl_xor(cm, 1, 32));
      cm = fmaxf(cm, __shfl_xor(cm, 2, 32));
      cm = fmaxf(cm, __shfl_xor(cm, 4, 32));
      cm = fmaxf(cm, __shfl_xor(cm, 8, 32));
      const float nm = fmaxf(m[i], cm);
      const float rescale = __expf(m[i] - nm);
      const float p0 = __expf(a0 - nm);
      const float p1 = __expf(a1 - nm);
      float rs = p0 + p1;
      rs += __shfl_xor(rs, 1, 32);
      rs += __shfl_xor(rs, 2, 32);
      rs += __shfl_xor(rs, 4, 32);
      rs += __shfl_xor(rs, 8, 32);
      l[i] = l[i] * rescale + rs;
      m[i] = nm;
#pragma unroll
      for (int n = 0; n < 4; ++n) o[n][i] *= rescale;
      const int r = rbase + i;                      // C-layout -> LDS tile
      P[r * 40 + col]      = (__bf16)p0;
      P[r * 40 + 16 + col] = (__bf16)p1;
    }
    asm volatile("s_wait_dscnt 0x0" ::: "memory");  // per-wave LDS RAW fence
    // reload P in A-fragment layout (16x32)
    v16bf pf;
    {
      const int r  = lane & 15;
      const int b0 = (lane < 16) ? 0 : 8;
#pragma unroll
      for (int j = 0; j < 8; ++j) pf[j]     = P[r * 40 + b0 + j];
#pragma unroll
      for (int j = 0; j < 8; ++j) pf[8 + j] = P[r * 40 + b0 + 16 + j];
    }
    // O += P @ V  (columns of B = rows of the transposed V tile in LDS)
    v16bf vf[4];
#pragma unroll
    for (int n = 0; n < 4; ++n) vf[n] = load_b_frag(vt, 40, n * 16, 0, lane);
#pragma unroll
    for (int n = 0; n < 4; ++n) o[n] = wmma_bf16(pf, vf[n], o[n]);
    __syncthreads();          // everyone done reading buf[cur] before rewrite
  }
#undef STAGE_TILES

  // normalize and store O as bf16 [t][h*64+d]
#pragma unroll
  for (int i = 0; i < 8; ++i) {
    const float inv = 1.0f / l[i];
    const int t = q0 + rbase + i;
#pragma unroll
    for (int n = 0; n < 4; ++n)
      O[(size_t)t * EMBED + head * HDIM + n * 16 + col] = (__bf16)(o[n][i] * inv);
  }
}

// ------------------------------------------------------- output projection
// Same LDS-staged block GEMM; fp32 epilogue straight to d_out.
__global__ void out_gemm(const __bf16* __restrict__ Ob,
                         const __bf16* __restrict__ wb,
                         float* __restrict__ out) {
  const int tid  = threadIdx.x;
  const int lane = tid & 31;
  const int wave = tid >> 5;
  const int NT   = EMBED / 128;                    // 8 n-tiles
  const int bt0  = (blockIdx.x / NT) * 128;
  const int f0   = (blockIdx.x % NT) * 128;
  const int t0   = bt0 + wave * 16;

  __shared__ __bf16 sA[2][128 * 72];
  __shared__ __bf16 sB[2][128 * 72];
  const unsigned aA[2] = { (unsigned)(size_t)&sA[0][0], (unsigned)(size_t)&sA[1][0] };
  const unsigned aB[2] = { (unsigned)(size_t)&sB[0][0], (unsigned)(size_t)&sB[1][0] };
  const unsigned long long gA = (unsigned long long)(size_t)Ob;
  const unsigned long long gB = (unsigned long long)(size_t)wb;

  v8f acc[8] = {};
  const int NSTEP = EMBED / 64;                    // 16 K-steps of 64
  stage_ab(gA, gB, bt0, f0, 0, aA[0], aB[0], tid);
  for (int it = 0; it < NSTEP; ++it) {
    const int cur = it & 1;
    const int nk  = (it + 1 < NSTEP) ? (it + 1) * 64 : it * 64;
    stage_ab(gA, gB, bt0, f0, nk, aA[cur ^ 1], aB[cur ^ 1], tid);
    asm volatile("s_wait_asynccnt 0x8" ::: "memory");
    __syncthreads();
    const __bf16* sa = &sA[cur][0];
    const __bf16* sb = &sB[cur][0];
#pragma unroll
    for (int kd = 0; kd < 2; ++kd) {
      v16bf a = load_a_frag(sa, 72, wave * 16, kd * 32, lane);
#pragma unroll
      for (int n = 0; n < 8; ++n) {
        v16bf b = load_b_frag(sb, 72, n * 16, kd * 32, lane);
        acc[n] = wmma_bf16(a, b, acc[n]);
      }
    }
    __syncthreads();
  }

  const int col   = lane & 15;
  const int rbase = (lane < 16) ? 0 : 8;
#pragma unroll
  for (int n = 0; n < 8; ++n)
#pragma unroll
    for (int i = 0; i < 8; ++i)
      out[(size_t)(t0 + rbase + i) * EMBED + f0 + n * 16 + col] = acc[n][i];
}

extern "C" void kernel_launch(void* const* d_in, const int* in_sizes, int n_in,
                              void* d_out, int out_size, void* d_ws, size_t ws_size,
                              hipStream_t stream) {
  const float* x     = (const float*)d_in[0];   // [1,4096,1024]
  const float* w_qkv = (const float*)d_in[1];   // [3072,1024]
  const float* w_out = (const float*)d_in[2];   // [1024,1024]
  float* out = (float*)d_out;

  __bf16* ws = (__bf16*)d_ws;
  size_t off = 0;
  __bf16* xb    = ws + off; off += (size_t)SEQ * EMBED;       // 4M
  __bf16* wqkvb = ws + off; off += (size_t)3 * EMBED * EMBED; // 3M
  __bf16* woutb = ws + off; off += (size_t)EMBED * EMBED;     // 1M
  __bf16* Q     = ws + off; off += (size_t)SEQ * EMBED;       // 4M
  __bf16* K     = ws + off; off += (size_t)SEQ * EMBED;       // 4M
  __bf16* Vt    = ws + off; off += (size_t)SEQ * EMBED;       // 4M
  __bf16* Ob    = ws + off; off += (size_t)SEQ * EMBED;       // 4M  (~48 MB)

  const int n_x  = SEQ * EMBED;
  const int n_wq = 3 * EMBED * EMBED;
  const int n_wo = EMBED * EMBED;
  cvt_f32_bf16<<<n_x  / 4 / 256, 256, 0, stream>>>(x,     xb,    n_x);
  cvt_f32_bf16<<<n_wq / 4 / 256, 256, 0, stream>>>(w_qkv, wqkvb, n_wq);
  cvt_f32_bf16<<<n_wo / 4 / 256, 256, 0, stream>>>(w_out, woutb, n_wo);

  // (4096/128 M-tiles) * (3072/128 N-tiles) blocks
  qkv_gemm<<<32 * 24, 256, 0, stream>>>(xb, wqkvb, Q, K, Vt);
  // 16 heads * 32 blocks (8 waves * 16 queries each)
  flash_attn<<<HEADS * 32, 256, 0, stream>>>(Q, K, Vt, Ob);
  // (4096/128) * (1024/128) blocks
  out_gemm<<<32 * 8, 256, 0, stream>>>(Ob, woutb, out);
}
